// ImageEncoderViT_33913061769927
// MI455X (gfx1250) — matrix operand order repaired and checked by reading, not compiled, not run on hardware
//
#include <hip/hip_runtime.h>
#include <hip/hip_bf16.h>
#include <math.h>

// ---------------------------------------------------------------------------
// CDNA5 / gfx1250 WMMA types and helpers (wave32)
// ---------------------------------------------------------------------------
typedef __attribute__((ext_vector_type(16))) __bf16 v16bf;
typedef __attribute__((ext_vector_type(8)))  float  v8f;
typedef __attribute__((ext_vector_type(4)))  unsigned u32x4;
typedef __attribute__((ext_vector_type(8)))  unsigned u32x8;

__device__ __forceinline__ v8f wmma_bf16(v16bf a, v16bf b, v8f c) {
  // D = A(16x32 bf16) * B(32x16 bf16) + C(16x16 f32)
  return __builtin_amdgcn_wmma_f32_16x16x32_bf16(
      /*neg_a=*/false, a, /*neg_b=*/false, b,
      /*c_mod=*/(short)0, c, /*reuse_a=*/false, /*reuse_b=*/false);
}

// LDS byte offset of a generic pointer into __shared__ (addrspacecast -> AS3).
__device__ __forceinline__ unsigned lds_off(const void* p) {
  return (unsigned)(unsigned long long)
      (__attribute__((address_space(3))) const char*)p;
}

// ---------------------------------------------------------------------------
// CDNA5 async global->LDS copies (ASYNCcnt-tracked; no VGPR round trip).
// VGLOBAL "GV" form: dst = LDS byte address in VGPR, src = 64-bit VGPR addr.
// ---------------------------------------------------------------------------
__device__ __forceinline__ void async_ld_b128(unsigned dst_lds, const void* src) {
  asm volatile("global_load_async_to_lds_b128 %0, %1, off"
               :: "v"(dst_lds), "v"((unsigned long long)(uintptr_t)src)
               : "memory");
}
__device__ __forceinline__ void async_ld_b64(unsigned dst_lds, const void* src) {
  asm volatile("global_load_async_to_lds_b64 %0, %1, off"
               :: "v"(dst_lds), "v"((unsigned long long)(uintptr_t)src)
               : "memory");
}
__device__ __forceinline__ void wait_async0() {
  asm volatile("s_wait_asynccnt 0x0" ::: "memory");
}

// ---------------------------------------------------------------------------
// Tensor Data Mover: 2D tile load (groups 2/3 disabled => <=2D tensor).
// Descriptor packing per CDNA5 ISA ch.8 (D# group0 128b, group1 256b).
// data_size = 2 bytes (bf16); dims/strides in 2B units; LDS row padding via
// pad_interval/pad_amount so the landed tile matches a padded LDS stride.
// ---------------------------------------------------------------------------
__device__ __forceinline__ void tdm_load_2d(unsigned lds_byte_off, const void* g,
                                            unsigned tile_d0,   // elems (2B)
                                            unsigned tile_d1,   // rows
                                            unsigned stride,    // elems (2B)
                                            unsigned pad_en,
                                            unsigned pad_interval_code,
                                            unsigned pad_amount_code) {
  unsigned long long ga = (unsigned long long)(uintptr_t)g;
  u32x4 g0;
  g0[0] = 1u;                                   // count=1, user descriptor
  g0[1] = lds_byte_off;                         // lds_addr [63:32]
  g0[2] = (unsigned)(ga & 0xffffffffu);         // global_addr[31:0]  -> [95:64]
  g0[3] = (unsigned)((ga >> 32) & 0x01ffffffu)  // global_addr[56:32] -> [120:96]
        | (2u << 30);                           // type=2 ("image")   -> [127:126]
  u32x8 g1;
  g1[0] = (1u << 16)                            // data_size: 1 => 2 bytes
        | (pad_en << 20)                        // pad_enable
        | (pad_interval_code << 22)             // [24:22]
        | (pad_amount_code << 25);              // [31:25]
  g1[1] = (tile_d0 & 0xffffu) << 16;            // tensor_dim0[15:0] (abar=0)
  g1[2] = (tile_d0 >> 16) | ((tile_d1 & 0xffffu) << 16);  // td0 hi | tensor_dim1 lo
  g1[3] = (tile_d1 >> 16) | ((tile_d0 & 0xffffu) << 16);  // td1 hi | tile_dim0
  g1[4] = tile_d1 & 0xffffu;                    // tile_dim1 (tile_dim2 = 0)
  g1[5] = stride;                               // tensor_dim0_stride[31:0]
  g1[6] = 0u;                                   // stride hi | dim1_stride lo
  g1[7] = 0u;
  asm volatile("tensor_load_to_lds %0, %1" :: "s"(g0), "s"(g1) : "memory");
}

// ---------------------------------------------------------------------------
// WMMA fragment loads from LDS (layouts per ISA 7.12.2, wave32)
// ---------------------------------------------------------------------------
__device__ __forceinline__ v16bf fragA_ld(const __bf16* lds, int stride,
                                          int r0, int k0, int lane) {
  int m  = lane & 15;
  int kb = (lane >> 4) << 3;           // 0 or 8
  const __bf16* row = lds + (r0 + m) * stride + k0;
  union { v16bf v; unsigned u[8]; } t;
#pragma unroll
  for (int j = 0; j < 8; ++j) {
    int k = ((j >> 2) << 4) + kb + ((j & 3) << 1);
    t.u[j] = *(const unsigned*)(row + k);
  }
  return t.v;
}
__device__ __forceinline__ v16bf fragB_ld(const __bf16* lds, int stride,
                                          int n0, int k0, int lane) {
  int n  = lane & 15;
  int kb = (lane >> 4) << 3;           // 0 or 8
  const __bf16* p = lds + (n0 + n) * stride + k0 + kb;
  union { v16bf v; uint4 u[2]; } t;
  t.u[0] = *(const uint4*)p;
  t.u[1] = *(const uint4*)(p + 16);
  return t.v;
}

// ---------------------------------------------------------------------------
// index maps: window-partition order <-> natural [B,16,16,16] token order
// ---------------------------------------------------------------------------
__device__ __forceinline__ int win2nat(int t) {
  int win = t >> 9, r = t & 511;
  int b  = win >> 3;
  int H  = (((win >> 2) & 1) << 3) | (r >> 6);
  int W  = (((win >> 1) & 1) << 3) | ((r >> 3) & 7);
  int Z  = ((win & 1) << 3)        | (r & 7);
  return ((b * 16 + H) * 16 + W) * 16 + Z;
}
__device__ __forceinline__ int nat2win(int t) {
  int b = t >> 12, H = (t >> 8) & 15, W = (t >> 4) & 15, Z = t & 15;
  int win = ((((b << 1) | (H >> 3)) << 1) | (W >> 3)) << 1 | (Z >> 3);
  int r   = ((((H & 7) << 3) | (W & 7)) << 3) | (Z & 7);
  return (win << 9) | r;
}

// ---------------------------------------------------------------------------
// fp32 -> bf16 bulk convert (weights, once per launch; cheap vs GEMMs)
// ---------------------------------------------------------------------------
__global__ void cvt_bf16_kernel(const float* __restrict__ in,
                                __bf16* __restrict__ out, int n) {
  for (int i = blockIdx.x * blockDim.x + threadIdx.x; i < n;
       i += gridDim.x * blockDim.x)
    out[i] = (__bf16)in[i];
}

// ---------------------------------------------------------------------------
// LayerNorm over C=768, fp32 in, bf16 out; optional natural->window reorder
// ---------------------------------------------------------------------------
__global__ __launch_bounds__(256)
void ln_kernel(const float* __restrict__ x, const float* __restrict__ g,
               const float* __restrict__ b, __bf16* __restrict__ out,
               int reorder) {
  __shared__ float rs[256], rq[256];
  int t = blockIdx.x, tid = threadIdx.x;
  const float* row = x + (size_t)t * 768;
  float v0 = row[tid], v1 = row[tid + 256], v2 = row[tid + 512];
  rs[tid] = v0 + v1 + v2;
  rq[tid] = v0 * v0 + v1 * v1 + v2 * v2;
  __syncthreads();
  for (int off = 128; off > 0; off >>= 1) {
    if (tid < off) { rs[tid] += rs[tid + off]; rq[tid] += rq[tid + off]; }
    __syncthreads();
  }
  float mu  = rs[0] * (1.f / 768.f);
  float var = rq[0] * (1.f / 768.f) - mu * mu;
  float ri  = rsqrtf(var + 1e-5f);
  int dst = reorder ? nat2win(t) : t;
  __bf16* orow = out + (size_t)dst * 768;
  orow[tid]       = (__bf16)((v0 - mu) * ri * g[tid]       + b[tid]);
  orow[tid + 256] = (__bf16)((v1 - mu) * ri * g[tid + 256] + b[tid + 256]);
  orow[tid + 512] = (__bf16)((v2 - mu) * ri * g[tid + 512] + b[tid + 512]);
}

// ---------------------------------------------------------------------------
// bf16 WMMA GEMM, double-buffered with async global->LDS A-tile staging.
// Block tile 128x64, BK=32, 256 threads = 8 waves (4x2), wave tile 32x32.
// EPI: 0 = +bias -> bf16 (QKV)
//      1 = +bias, window->natural, +resid(x) -> f32 (proj)
//      2 = +bias, exact GELU -> bf16 (mlp fc1)
//      3 = +bias, +resid(h) -> f32 (mlp fc2 -> d_out)
// ---------------------------------------------------------------------------
#define GM_BM 128
#define GM_BN 64
#define GM_BK 32
#define GM_SA 40   // 32 + 8 pad (keeps 16B LDS alignment)
#define GM_SB 40

template <int EPI>
__global__ __launch_bounds__(256)
void gemm_bf16_kernel(const __bf16* __restrict__ A, const __bf16* __restrict__ B,
                      int M, int N, int K,
                      const float* __restrict__ bias,
                      const float* __restrict__ resid,
                      float* __restrict__ outf,
                      __bf16* __restrict__ outb) {
  __shared__ __bf16 sA[2][GM_BM * GM_SA];
  __shared__ __bf16 sBt[2][GM_BN * GM_SB];

  int tid = threadIdx.x, lane = tid & 31, wv = tid >> 5;
  int wy = wv >> 1, wx = wv & 1;              // 4x2 wave grid
  int m0 = blockIdx.y * GM_BM, n0 = blockIdx.x * GM_BN;
  int hb = (lane >> 4) << 3, ln = lane & 15;

  v8f acc[2][2];
  v8f zro = {};
  acc[0][0] = zro; acc[0][1] = zro; acc[1][0] = zro; acc[1][1] = zro;

  const __bf16* Ab = A + (size_t)m0 * K;
  int ar = tid >> 1, ac = (tid & 1) << 4;     // A-tile: row, 16-col segment

  // A tile via async DMA (2x b128 per thread), B tile transposed manually.
  auto stageA = [&](int buf, int kb) {
    const __bf16* src = Ab + (size_t)ar * K + kb + ac;
    async_ld_b128(lds_off(&sA[buf][ar * GM_SA + ac]), src);
    async_ld_b128(lds_off(&sA[buf][ar * GM_SA + ac + 8]), src + 8);
  };
  auto stageB = [&](int buf, int kb) {
#pragma unroll
    for (int it = 0; it < 8; ++it) {
      int idx = tid + (it << 8);
      int k = idx >> 6, n = idx & 63;
      sBt[buf][n * GM_SB + k] = B[(size_t)(kb + k) * N + n0 + n];
      if (it == 0 && kb + GM_BK < K)
        __builtin_prefetch(&B[(size_t)(kb + GM_BK + k) * N + n0 + n], 0, 0);
    }
  };

  stageA(0, 0);
  stageB(0, 0);
  wait_async0();
  __syncthreads();

  int cur = 0;
  for (int kb = 0; kb < K; kb += GM_BK) {
    if (kb + GM_BK < K) {                     // prefetch next tile into alt buf
      stageA(cur ^ 1, kb + GM_BK);
      stageB(cur ^ 1, kb + GM_BK);
    }
    const __bf16* cA = sA[cur];
    const __bf16* cB = sBt[cur];
    v16bf a0 = fragA_ld(cA, GM_SA, wy * 32,      0, lane);
    v16bf a1 = fragA_ld(cA, GM_SA, wy * 32 + 16, 0, lane);
    v16bf b0 = fragB_ld(cB, GM_SB, wx * 32,      0, lane);
    v16bf b1 = fragB_ld(cB, GM_SB, wx * 32 + 16, 0, lane);
    acc[0][0] = wmma_bf16(a0, b0, acc[0][0]);
    acc[0][1] = wmma_bf16(a0, b1, acc[0][1]);
    acc[1][0] = wmma_bf16(a1, b0, acc[1][0]);
    acc[1][1] = wmma_bf16(a1, b1, acc[1][1]);
    wait_async0();                            // next-buf DMA landed
    __syncthreads();
    cur ^= 1;
  }

#pragma unroll
  for (int mt = 0; mt < 2; ++mt)
#pragma unroll
    for (int nt = 0; nt < 2; ++nt)
#pragma unroll
      for (int i = 0; i < 8; ++i) {
        int m = m0 + wy * 32 + mt * 16 + i + hb;
        int n = n0 + wx * 32 + nt * 16 + ln;
        float v = acc[mt][nt][i] + bias[n];
        if (EPI == 0) {
          outb[(size_t)m * N + n] = (__bf16)v;
        } else if (EPI == 1) {
          int t = win2nat(m);
          size_t o = (size_t)t * 768 + n;
          outf[o] = resid[o] + v;
        } else if (EPI == 2) {
          float gl = 0.5f * v * (1.f + erff(v * 0.70710678118f));
          outb[(size_t)m * N + n] = (__bf16)gl;
        } else {
          size_t o = (size_t)m * N + n;
          outf[o] = resid[o] + v;
        }
      }
}

// ---------------------------------------------------------------------------
// Windowed attention, flash-style streaming softmax.
// 1 WG = (window, head, 128-query tile); 8 waves x 16 query rows each.
// Q staged via async b64 DMA; K chunk staged by the Tensor Data Mover
// (64 rows x 128B, tensor stride 4608B, LDS pad 4 DWORDs per 32 => AT_STR=72).
// ---------------------------------------------------------------------------
#define AT_STR 72  // 64 + 8 pad
__global__ __launch_bounds__(256)
void attn_kernel(const __bf16* __restrict__ qkv,
                 const float* __restrict__ relh,
                 const float* __restrict__ relw,
                 const float* __restrict__ relz,
                 __bf16* __restrict__ out) {
  __shared__ __bf16 sQ[128 * AT_STR];
  __shared__ __bf16 sK[64 * AT_STR];
  __shared__ __bf16 sVt[64 * AT_STR];         // V transposed: [c][key]
  __shared__ __bf16 sP[8 * 16 * AT_STR];      // per-wave P tiles
  __shared__ float  sRel[128 * 24];

  int tid = threadIdx.x, lane = tid & 31, wv = tid >> 5;
  int hb = (lane >> 4) << 3, ln = lane & 15;
  int qb   = blockIdx.x & 3;                  // q tile (128 rows)
  int wh   = blockIdx.x >> 2;
  int win  = wh / 12, head = wh % 12;
  const size_t tokbase = (size_t)win * 512;
  const int QS = 2304;

  // stage Q (unscaled): 128 rows x 64 cols, async b64 per 8B chunk
#pragma unroll
  for (int it = 0; it < 8; ++it) {
    int cid = tid + (it << 8);
    int r = cid >> 4, c4 = (cid & 15) << 2;
    const __bf16* src = qkv + (tokbase + qb * 128 + r) * QS + head * 64 + c4;
    async_ld_b64(lds_off(&sQ[r * AT_STR + c4]), src);
  }
  wait_async0();
  __syncthreads();

  // decomposed rel-pos bias components: sRel[r][0..7]=h, [8..15]=w, [16..23]=z
  for (int idx = tid; idx < 128 * 24; idx += 256) {
    int r = idx / 24, j = idx - r * 24;
    int qg = qb * 128 + r;
    int hq = qg >> 6, wq = (qg >> 3) & 7, zq = qg & 7;
    const float* tab; int coord;
    if (j < 8)       { tab = relh; coord = hq - j + 7; }
    else if (j < 16) { tab = relw; coord = wq - (j - 8) + 7; }
    else             { tab = relz; coord = zq - (j - 16) + 7; }
    const float* rp = tab + coord * 64;
    float s = 0.f;
#pragma unroll
    for (int c = 0; c < 64; ++c) s += (float)sQ[r * AT_STR + c] * rp[c];
    sRel[idx] = s;
  }

  // per-wave Q fragments (rows wv*16..+15, K halves 0 and 32)
  v16bf qa0 = fragA_ld(sQ, AT_STR, wv * 16, 0, lane);
  v16bf qa1 = fragA_ld(sQ, AT_STR, wv * 16, 32, lane);

  v8f oa[4];
  v8f zro = {};
#pragma unroll
  for (int nt = 0; nt < 4; ++nt) oa[nt] = zro;
  float rm[8], rl[8];
#pragma unroll
  for (int i = 0; i < 8; ++i) { rm[i] = -1e30f; rl[i] = 0.f; }
  __bf16* pb = sP + wv * 16 * AT_STR;

  for (int ck = 0; ck < 8; ++ck) {            // 8 chunks of 64 keys
    __syncthreads();
    // K chunk [key][c] via Tensor Data Mover (one issue per workgroup):
    // 64 elems/row (2B), 64 rows, stride 2304 elems; LDS pad: every 32 DWORDs
    // (code 4) insert 4 DWORDs (code 3) -> row pitch 72 bf16 == AT_STR.
    if (wv == 0) {
      const __bf16* kg = qkv + (tokbase + ck * 64) * QS + 768 + head * 64;
      tdm_load_2d(lds_off(&sK[0]), kg, 64u, 64u, (unsigned)QS, 1u, 4u, 3u);
    }
    // V chunk transposed [c][key] (manual: TDM cannot transpose 2B elements)
#pragma unroll
    for (int it = 0; it < 16; ++it) {
      int e = tid + (it << 8);
      int k = e >> 6, c = e & 63;
      sVt[c * AT_STR + k] = qkv[(tokbase + ck * 64 + k) * QS + 1536 + head * 64 + c];
    }
    if (wv == 0) __builtin_amdgcn_s_wait_tensorcnt(0);
    __syncthreads();

    // S tile = scale * Q K^T + bias  (16 rows x 64 keys per wave)
    float sv[4][8];
#pragma unroll
    for (int nt = 0; nt < 4; ++nt) {
      v8f a = zro;
      a = wmma_bf16(qa0, fragB_ld(sK, AT_STR, nt * 16, 0, lane), a);
      a = wmma_bf16(qa1, fragB_ld(sK, AT_STR, nt * 16, 32, lane), a);
      int kg = ck * 64 + nt * 16 + ln;
      int kh = kg >> 6, kw = (kg >> 3) & 7, kz = kg & 7;
#pragma unroll
      for (int i = 0; i < 8; ++i) {
        int r = wv * 16 + i + hb;
        sv[nt][i] = a[i] * 0.125f + sRel[r * 24 + kh]
                  + sRel[r * 24 + 8 + kw] + sRel[r * 24 + 16 + kz];
      }
    }

    // online softmax: each row lives entirely in one 16-lane half
    float cm[8];
#pragma unroll
    for (int i = 0; i < 8; ++i)
      cm[i] = fmaxf(fmaxf(sv[0][i], sv[1][i]), fmaxf(sv[2][i], sv[3][i]));
#pragma unroll
    for (int m = 1; m < 16; m <<= 1)
#pragma unroll
      for (int i = 0; i < 8; ++i) cm[i] = fmaxf(cm[i], __shfl_xor(cm[i], m, 32));
    float al[8], ps[8];
#pragma unroll
    for (int i = 0; i < 8; ++i) {
      float nm = fmaxf(rm[i], cm[i]);
      al[i] = __expf(rm[i] - nm);
      rm[i] = nm;
      ps[i] = 0.f;
    }
#pragma unroll
    for (int nt = 0; nt < 4; ++nt)
#pragma unroll
      for (int i = 0; i < 8; ++i) {
        float p = __expf(sv[nt][i] - rm[i]);
        sv[nt][i] = p;
        ps[i] += p;
      }
#pragma unroll
    for (int m = 1; m < 16; m <<= 1)
#pragma unroll
      for (int i = 0; i < 8; ++i) ps[i] += __shfl_xor(ps[i], m, 32);
#pragma unroll
    for (int i = 0; i < 8; ++i) rl[i] = rl[i] * al[i] + ps[i];
#pragma unroll
    for (int nt = 0; nt < 4; ++nt)
#pragma unroll
      for (int i = 0; i < 8; ++i) oa[nt][i] = oa[nt][i] * al[i];

    // repack P (C-layout f32 -> A-layout bf16) via wave-private LDS
#pragma unroll
    for (int nt = 0; nt < 4; ++nt)
#pragma unroll
      for (int i = 0; i < 8; ++i)
        pb[(i + hb) * AT_STR + nt * 16 + ln] = (__bf16)sv[nt][i];
    v16bf pa0 = fragA_ld(pb, AT_STR, 0, 0, lane);
    v16bf pa1 = fragA_ld(pb, AT_STR, 0, 32, lane);
#pragma unroll
    for (int nt = 0; nt < 4; ++nt) {
      oa[nt] = wmma_bf16(pa0, fragB_ld(sVt, AT_STR, nt * 16, 0, lane), oa[nt]);
      oa[nt] = wmma_bf16(pa1, fragB_ld(sVt, AT_STR, nt * 16, 32, lane), oa[nt]);
    }
  }

  // normalize and store (window-token order, [tok][head*64+c], bf16)
#pragma unroll
  for (int nt = 0; nt < 4; ++nt)
#pragma unroll
    for (int i = 0; i < 8; ++i) {
      float o = oa[nt][i] / rl[i];
      int q = qb * 128 + wv * 16 + i + hb;
      int c = nt * 16 + ln;
      out[(tokbase + q) * 768 + head * 64 + c] = (__bf16)o;
    }
}

// ---------------------------------------------------------------------------
// launch
// ---------------------------------------------------------------------------
extern "C" void kernel_launch(void* const* d_in, const int* in_sizes, int n_in,
                              void* d_out, int out_size, void* d_ws, size_t ws_size,
                              hipStream_t stream) {
  (void)in_sizes; (void)n_in; (void)out_size; (void)ws_size;
  const float* x      = (const float*)d_in[0];
  const float* w_qkv  = (const float*)d_in[1];
  const float* b_qkv  = (const float*)d_in[2];
  const float* w_proj = (const float*)d_in[3];
  const float* b_proj = (const float*)d_in[4];
  const float* relh   = (const float*)d_in[5];
  const float* relw   = (const float*)d_in[6];
  const float* relz   = (const float*)d_in[7];
  const float* ln1g   = (const float*)d_in[8];
  const float* ln1b   = (const float*)d_in[9];
  const float* ln2g   = (const float*)d_in[10];
  const float* ln2b   = (const float*)d_in[11];
  const float* w1     = (const float*)d_in[12];
  const float* b1     = (const float*)d_in[13];
  const float* w2     = (const float*)d_in[14];
  const float* b2     = (const float*)d_in[15];
  float* out = (float*)d_out;
  char*  ws  = (char*)d_ws;

  // workspace layout (bytes)
  constexpr size_t SZ_WQKV  = 768ull * 2304 * 2;
  constexpr size_t SZ_WPROJ = 768ull * 768 * 2;
  constexpr size_t SZ_W1    = 768ull * 3072 * 2;
  constexpr size_t SZ_W2    = 3072ull * 768 * 2;
  constexpr size_t SZ_LN    = 8192ull * 768 * 2;
  constexpr size_t SZ_QKV   = 8192ull * 2304 * 2;
  constexpr size_t SZ_H     = 8192ull * 768 * 4;
  constexpr size_t SZ_M1    = 8192ull * 3072 * 2;
  size_t o = 0;
  __bf16* wqkv_bf  = (__bf16*)(ws + o); o += SZ_WQKV;
  __bf16* wproj_bf = (__bf16*)(ws + o); o += SZ_WPROJ;
  __bf16* w1_bf    = (__bf16*)(ws + o); o += SZ_W1;
  __bf16* w2_bf    = (__bf16*)(ws + o); o += SZ_W2;
  __bf16* ln1w     = (__bf16*)(ws + o); o += SZ_LN;   // window-ordered LN1 out
  __bf16* qkvb     = (__bf16*)(ws + o); o += SZ_QKV;  // [tok][q|k|v]
  __bf16* attnb    = (__bf16*)(ws + o); o += SZ_LN;   // attention out (win order)
  float*  hbuf     = (float*)(ws + o);  o += SZ_H;    // h = x + attn (natural)
  __bf16* mlnb     = (__bf16*)(ws + o); o += SZ_LN;   // LN2 out (natural)
  __bf16* m1b      = (__bf16*)(ws + o); o += SZ_M1;   // gelu(fc1)

  // 1) weights -> bf16
  cvt_bf16_kernel<<<1024, 256, 0, stream>>>(w_qkv,  wqkv_bf,  768 * 2304);
  cvt_bf16_kernel<<<1024, 256, 0, stream>>>(w_proj, wproj_bf, 768 * 768);
  cvt_bf16_kernel<<<1024, 256, 0, stream>>>(w1,     w1_bf,    768 * 3072);
  cvt_bf16_kernel<<<1024, 256, 0, stream>>>(w2,     w2_bf,    3072 * 768);

  // 2) LN1 + window partition
  ln_kernel<<<8192, 256, 0, stream>>>(x, ln1g, ln1b, ln1w, 1);

  // 3) QKV GEMM (8192x2304x768), fused +bias -> bf16
  gemm_bf16_kernel<0><<<dim3(36, 64), 256, 0, stream>>>(
      ln1w, wqkv_bf, 8192, 2304, 768, b_qkv, nullptr, nullptr, qkvb);

  // 4) windowed attention w/ rel-pos bias (192 window-heads x 4 q-tiles)
  attn_kernel<<<768, 256, 0, stream>>>(qkvb, relh, relw, relz, attnb);

  // 5) proj GEMM, fused +bias + window-unpartition + residual(x) -> h (f32)
  gemm_bf16_kernel<1><<<dim3(12, 64), 256, 0, stream>>>(
      attnb, wproj_bf, 8192, 768, 768, b_proj, x, hbuf, nullptr);

  // 6) LN2
  ln_kernel<<<8192, 256, 0, stream>>>(hbuf, ln2g, ln2b, mlnb, 0);

  // 7) fc1 GEMM, fused +bias + exact GELU -> bf16
  gemm_bf16_kernel<2><<<dim3(48, 64), 256, 0, stream>>>(
      mlnb, w1_bf, 8192, 3072, 768, b1, nullptr, nullptr, m1b);

  // 8) fc2 GEMM, fused +bias + residual(h) -> d_out (f32)
  gemm_bf16_kernel<3><<<dim3(12, 64), 256, 0, stream>>>(
      m1b, w2_bf, 8192, 768, 3072, b2, hbuf, out, nullptr);
}
